// MultiHeadedDotAttention_17282948399482
// MI455X (gfx1250) — matrix operand order repaired and checked by reading, not compiled, not run on hardware
//
#include <hip/hip_runtime.h>
#include <hip/hip_bf16.h>
#include <math.h>

// ---------------- problem constants (match reference) ----------------
#define Bb   16
#define Tt   1024
#define Dd   1024
#define Hh   8
#define DKk  128          // Dd / Hh
#define D2   2048         // 2*Dd

typedef __bf16 bf16;
typedef bf16  v16bf __attribute__((ext_vector_type(16)));
typedef bf16  v8bf  __attribute__((ext_vector_type(8)));
typedef float v8f   __attribute__((ext_vector_type(8)));
typedef float v4f   __attribute__((ext_vector_type(4)));

static __device__ __forceinline__ v8f zero8() {
  v8f z;
#pragma unroll
  for (int i = 0; i < 8; ++i) z[i] = 0.0f;
  return z;
}

// ---- A-matrix fragment (16x32, bf16): lane holds row M=lane%16.
// g = lane/16.  halfs[0..7] = K = g*8..g*8+7 ; halfs[8..15] = K = 16+g*8..+7
static __device__ __forceinline__ v16bf a_frag_bf16(const bf16* __restrict__ row,
                                                    int kb, int g) {
  const v8bf lo = *(const v8bf*)(row + kb + g * 8);
  const v8bf hi = *(const v8bf*)(row + kb + 16 + g * 8);
  v16bf f;
#pragma unroll
  for (int i = 0; i < 8; ++i) { f[i] = lo[i]; f[8 + i] = hi[i]; }
  return f;
}

static __device__ __forceinline__ v16bf a_frag_f32(const float* __restrict__ row,
                                                   int kb, int g) {
  const v4f a0 = *(const v4f*)(row + kb + g * 8);
  const v4f a1 = *(const v4f*)(row + kb + g * 8 + 4);
  const v4f b0 = *(const v4f*)(row + kb + 16 + g * 8);
  const v4f b1 = *(const v4f*)(row + kb + 16 + g * 8 + 4);
  v16bf f;
#pragma unroll
  for (int i = 0; i < 4; ++i) {
    f[i]      = (bf16)a0[i];
    f[4 + i]  = (bf16)a1[i];
    f[8 + i]  = (bf16)b0[i];
    f[12 + i] = (bf16)b1[i];
  }
  return f;
}

// ---- B-matrix fragment (32x16, bf16): lane holds col N=lane%16.
// g = lane/16.  halfs[0..7] = K = g*8..g*8+7 ; halfs[8..15] = K = 16+g*8..+7
static __device__ __forceinline__ v16bf b_frag_bf16(const bf16* __restrict__ row,
                                                    int kb, int g) {
  const v8bf lo = *(const v8bf*)(row + kb + g * 8);
  const v8bf hi = *(const v8bf*)(row + kb + 16 + g * 8);
  v16bf f;
#pragma unroll
  for (int i = 0; i < 8; ++i) { f[i] = lo[i]; f[8 + i] = hi[i]; }
  return f;
}

static __device__ __forceinline__ v16bf b_frag_f32(const float* __restrict__ row,
                                                   int kb, int g) {
  v16bf f;
#pragma unroll
  for (int j = 0; j < 4; ++j) {
    const v4f v = *(const v4f*)(row + kb + g * 8 + (j >> 1) * 16 + (j & 1) * 4);
#pragma unroll
    for (int i = 0; i < 4; ++i) f[j * 4 + i] = (bf16)v[i];
  }
  return f;
}

static __device__ __forceinline__ v8f wmma_bf16(v16bf a, v16bf b, v8f c) {
  return __builtin_amdgcn_wmma_f32_16x16x32_bf16(false, a, false, b,
                                                 (short)0, c, false, false);
}

// =====================================================================
// Kernel 1: AoANet LayerNorm (ddof=1 std, eps on std), fp32 -> bf16 qn
// one block (256 thr) per row of 1024
// =====================================================================
__global__ __launch_bounds__(256) void ln_kernel(const float* __restrict__ x,
                                                 const float* __restrict__ gamma,
                                                 const float* __restrict__ beta,
                                                 bf16* __restrict__ qn) {
  __shared__ float s1[256], s2[256];
  const int row = blockIdx.x;
  const int tid = threadIdx.x;
  const float* xr = x + (size_t)row * Dd;
  float vals[4];
  float a = 0.f, b = 0.f;
#pragma unroll
  for (int i = 0; i < 4; ++i) {
    float v = xr[tid + i * 256];
    vals[i] = v;
    a += v;
    b += v * v;
  }
  s1[tid] = a; s2[tid] = b;
  __syncthreads();
  for (int off = 128; off > 0; off >>= 1) {
    if (tid < off) { s1[tid] += s1[tid + off]; s2[tid] += s2[tid + off]; }
    __syncthreads();
  }
  const float mu  = s1[0] * (1.0f / Dd);
  const float var = (s2[0] - (float)Dd * mu * mu) * (1.0f / (Dd - 1));
  const float inv = 1.0f / (sqrtf(fmaxf(var, 0.0f)) + 1e-6f);
  bf16* qr = qn + (size_t)row * Dd;
#pragma unroll
  for (int i = 0; i < 4; ++i) {
    int c = tid + i * 256;
    qr[c] = (bf16)(gamma[c] * (vals[i] - mu) * inv + beta[c]);
  }
}

// =====================================================================
// Kernel 2: projection GEMM  y = X @ W^T + bias
//   One wave computes a 16x64 output strip (4 N-tiles): one A-fragment
//   amortized over 4 WMMAs per K-step -> 1.25 frag loads per WMMA.
//   USE_BF16: A operand is bf16 (qn) vs fp32 converted on the fly.
//   MODE 0: out[b][h][t][d]  -- q, k      MODE 1: out[b][h][d][t] -- vT
// =====================================================================
template <int USE_BF16, int MODE>
__global__ __launch_bounds__(256) void proj_kernel(const bf16* __restrict__ Abf,
                                                   const float* __restrict__ Af32,
                                                   const float* __restrict__ W,
                                                   const float* __restrict__ bias,
                                                   bf16* __restrict__ out) {
  const int lane = threadIdx.x & 31;
  const int wave = blockIdx.x * (blockDim.x >> 5) + (threadIdx.x >> 5);
  const int NG = Dd / 64;                       // 16 n-groups of 4 tiles
  const int mtile = wave / NG, ngrp = wave % NG;
  const int g = lane >> 4, l16 = lane & 15;
  const int mA = mtile * 16 + l16;              // A row this lane loads

  const bf16*  arow_bf = Abf  + (size_t)mA * Dd;
  const float* arow_f  = Af32 + (size_t)mA * Dd;
  const float* wrow[4];
#pragma unroll
  for (int s = 0; s < 4; ++s)
    wrow[s] = W + (size_t)(ngrp * 64 + s * 16 + l16) * Dd;

  v8f acc[4];
#pragma unroll
  for (int s = 0; s < 4; ++s) acc[s] = zero8();

  for (int kb = 0; kb < Dd; kb += 32) {
    const v16bf a = USE_BF16 ? a_frag_bf16(arow_bf, kb, g)
                             : a_frag_f32(arow_f, kb, g);
#pragma unroll
    for (int s = 0; s < 4; ++s)
      acc[s] = wmma_bf16(a, b_frag_f32(wrow[s], kb, g), acc[s]);
  }

#pragma unroll
  for (int s = 0; s < 4; ++s) {
    const int nB = ngrp * 64 + s * 16 + l16;    // output col this lane owns
    const float bn = bias[nB];
    const int h = nB >> 7;                      // nB / DKk
    const int dd = nB & 127;                    // nB % DKk
#pragma unroll
    for (int r = 0; r < 8; ++r) {
      const int m = mtile * 16 + r + 8 * g;     // global output row
      const int bb = m >> 10, t = m & 1023;
      const float v = acc[s][r] + bn;
      if (MODE == 0)
        out[(((size_t)(bb * Hh + h) * Tt + t) * DKk) + dd] = (bf16)v;
      else
        out[(((size_t)(bb * Hh + h) * DKk + dd) * Tt) + t] = (bf16)v;
    }
  }
}

// =====================================================================
// Kernel 3: fused flash attention.  One wave = 16 query rows x dk=128.
//   scores = q k^T / sqrt(dk), online softmax, O = P V (via vT)
//   x output written [B,T,D] (col = h*dk+d) bf16 for the AoA GEMM.
// =====================================================================
__global__ __launch_bounds__(256) void attn_kernel(const bf16* __restrict__ qh,
                                                   const bf16* __restrict__ kh,
                                                   const bf16* __restrict__ vT,
                                                   bf16* __restrict__ xb) {
  __shared__ bf16 plds[8][16 * 32];             // per-wave P staging (1KB each)
  const int lane = threadIdx.x & 31;
  const int wloc = threadIdx.x >> 5;
  const int wave = blockIdx.x * 8 + wloc;
  const int bh = wave >> 6;                     // 64 m-tiles per (b,h)
  const int mt = wave & 63;
  const int g = lane >> 4, l16 = lane & 15;

  const bf16* qrow  = qh + ((size_t)bh * Tt + mt * 16 + l16) * DKk;
  const bf16* kbase = kh + (size_t)bh * Tt * DKk;
  const bf16* vbase = vT + (size_t)bh * DKk * Tt;

  // preload 16x128 q tile as 4 A-fragments (stays in VGPRs)
  v16bf qf[4];
#pragma unroll
  for (int f = 0; f < 4; ++f) qf[f] = a_frag_bf16(qrow, f * 32, g);

  v8f o[8];
#pragma unroll
  for (int dt = 0; dt < 8; ++dt) o[dt] = zero8();
  float mrun[8], lrun[8];
#pragma unroll
  for (int r = 0; r < 8; ++r) { mrun[r] = -INFINITY; lrun[r] = 0.0f; }

  const float scale = 0.08838834764831845f;     // 1/sqrt(128)

  for (int kc = 0; kc < Tt; kc += 32) {
    // ---- S = q k^T for 32 keys: two 16x16 C tiles, K=128 in 4 steps
    v8f s0 = zero8(), s1 = zero8();
    {
      const bf16* kr0 = kbase + (size_t)(kc + l16) * DKk;
      const bf16* kr1 = kbase + (size_t)(kc + 16 + l16) * DKk;
#pragma unroll
      for (int f = 0; f < 4; ++f) {
        s0 = wmma_bf16(qf[f], b_frag_bf16(kr0, f * 32, g), s0);
        s1 = wmma_bf16(qf[f], b_frag_bf16(kr1, f * 32, g), s1);
      }
    }

    // ---- online softmax (row m = r + 8*g lives in a 16-lane half)
#pragma unroll
    for (int r = 0; r < 8; ++r) {
      float a = s0[r] * scale, b = s1[r] * scale;
      float mr = fmaxf(a, b);
#pragma unroll
      for (int mk = 1; mk < 16; mk <<= 1)
        mr = fmaxf(mr, __shfl_xor(mr, mk, 32));
      const float mnew = fmaxf(mrun[r], mr);
      const float resc = __expf(mrun[r] - mnew);
      mrun[r] = mnew;
      lrun[r] *= resc;
#pragma unroll
      for (int dt = 0; dt < 8; ++dt) o[dt][r] *= resc;
      a = __expf(a - mnew);
      b = __expf(b - mnew);
      s0[r] = a; s1[r] = b;
      float rs = a + b;
#pragma unroll
      for (int mk = 1; mk < 16; mk <<= 1)
        rs += __shfl_xor(rs, mk, 32);
      lrun[r] += rs;
    }

    // ---- re-layout P: C-frag -> A-frag via LDS (same-wave, in-order DS)
    bf16* pw = plds[wloc];
#pragma unroll
    for (int r = 0; r < 8; ++r) {
      pw[(r + 8 * g) * 32 + l16]      = (bf16)s0[r];
      pw[(r + 8 * g) * 32 + 16 + l16] = (bf16)s1[r];
    }
    asm volatile("s_wait_dscnt 0" ::: "memory");
    const v16bf pf = a_frag_bf16(pw + (size_t)l16 * 32, 0, g);

    // ---- O += P @ V  (B-frags contiguous from vT rows)
#pragma unroll
    for (int dt = 0; dt < 8; ++dt) {
      const bf16* vrow = vbase + (size_t)(dt * 16 + l16) * Tt;
      o[dt] = wmma_bf16(pf, b_frag_bf16(vrow, kc, g), o[dt]);
    }
  }

  // ---- normalize and store x tile
  const int b = bh >> 3, h = bh & 7;
#pragma unroll
  for (int dt = 0; dt < 8; ++dt) {
    const int col = h * DKk + dt * 16 + l16;
#pragma unroll
    for (int r = 0; r < 8; ++r) {
      const int t = mt * 16 + r + 8 * g;
      xb[((size_t)b * Tt + t) * Dd + col] = (bf16)(o[dt][r] / lrun[r]);
    }
  }
}

// =====================================================================
// Kernel 4: AoA linear (2048->2048) + GLU, fused.
//   z = cat([x, qn]) @ W_aoa^T + b ; out = z[:, :D] * sigmoid(z[:, D:])
//   One wave computes the 'a' tile and its paired 'g' tile; the concat
//   K-dimension is handled as two straight-line loops (no branches).
// =====================================================================
__global__ __launch_bounds__(256) void aoa_kernel(const bf16* __restrict__ xb,
                                                  const bf16* __restrict__ qn,
                                                  const float* __restrict__ Waoa,
                                                  const float* __restrict__ baoa,
                                                  float* __restrict__ out) {
  const int lane = threadIdx.x & 31;
  const int wave = blockIdx.x * (blockDim.x >> 5) + (threadIdx.x >> 5);
  const int NT = Dd / 16;                        // 64 column tiles of the 'a' half
  const int mtile = wave / NT, ntile = wave % NT;
  const int g = lane >> 4, l16 = lane & 15;
  const int mA = mtile * 16 + l16;
  const int na = ntile * 16 + l16;
  const int ng = na + Dd;

  const bf16* xrow = xb + (size_t)mA * Dd;
  const bf16* qrow = qn + (size_t)mA * Dd;
  const float* wa = Waoa + (size_t)na * D2;
  const float* wg = Waoa + (size_t)ng * D2;

  v8f acc_a = zero8(), acc_g = zero8();
  // first half of K: A rows come from x
  for (int kb = 0; kb < Dd; kb += 32) {
    const v16bf a = a_frag_bf16(xrow, kb, g);
    acc_a = wmma_bf16(a, b_frag_f32(wa, kb, g), acc_a);
    acc_g = wmma_bf16(a, b_frag_f32(wg, kb, g), acc_g);
  }
  // second half of K: A rows come from qn
  for (int kb = 0; kb < Dd; kb += 32) {
    const v16bf a = a_frag_bf16(qrow, kb, g);
    acc_a = wmma_bf16(a, b_frag_f32(wa, Dd + kb, g), acc_a);
    acc_g = wmma_bf16(a, b_frag_f32(wg, Dd + kb, g), acc_g);
  }

  const float bna = baoa[na], bng = baoa[ng];
#pragma unroll
  for (int r = 0; r < 8; ++r) {
    const int m = mtile * 16 + r + 8 * g;
    const float av = acc_a[r] + bna;
    const float gv = acc_g[r] + bng;
    out[(size_t)m * Dd + na] = av * (1.0f / (1.0f + __expf(-gv)));
  }
}

// =====================================================================
extern "C" void kernel_launch(void* const* d_in, const int* in_sizes, int n_in,
                              void* d_out, int out_size, void* d_ws, size_t ws_size,
                              hipStream_t stream) {
  const float* query = (const float*)d_in[0];
  const float* key_  = (const float*)d_in[1];
  const float* value = (const float*)d_in[2];
  const float* Wq    = (const float*)d_in[3];
  const float* bq    = (const float*)d_in[4];
  const float* Wk    = (const float*)d_in[5];
  const float* bk    = (const float*)d_in[6];
  const float* Wv    = (const float*)d_in[7];
  const float* bv    = (const float*)d_in[8];
  const float* Waoa  = (const float*)d_in[9];
  const float* baoa  = (const float*)d_in[10];
  const float* ln_g  = (const float*)d_in[11];
  const float* ln_b  = (const float*)d_in[12];
  float* out = (float*)d_out;

  // workspace: 5 x 32MB bf16 tensors
  char* ws = (char*)d_ws;
  const size_t SZ = (size_t)Bb * Tt * Dd * sizeof(bf16);   // 32 MB
  bf16* qn = (bf16*)(ws);
  bf16* qh = (bf16*)(ws + SZ);
  bf16* kh = (bf16*)(ws + 2 * SZ);
  bf16* vT = (bf16*)(ws + 3 * SZ);
  bf16* xb = (bf16*)(ws + 4 * SZ);

  // 1. layernorm(query) -> qn (bf16)
  ln_kernel<<<Bb * Tt, 256, 0, stream>>>(query, ln_g, ln_b, qn);

  // 2. projections: (B*T/16)*(D/64) waves, 8 waves/block
  const int gemm_blocks = (Bb * Tt / 16) * (Dd / 64) / 8;  // 2048
  proj_kernel<1, 0><<<gemm_blocks, 256, 0, stream>>>(qn, nullptr, Wq, bq, qh);
  proj_kernel<0, 0><<<gemm_blocks, 256, 0, stream>>>(nullptr, key_, Wk, bk, kh);
  proj_kernel<0, 1><<<gemm_blocks, 256, 0, stream>>>(nullptr, value, Wv, bv, vT);

  // 3. fused attention: B*H*(T/16) waves / 8 per block
  const int attn_blocks = (Bb * Hh * (Tt / 16)) / 8;       // 1024
  attn_kernel<<<attn_blocks, 256, 0, stream>>>(qh, kh, vT, xb);

  // 4. AoA linear + GLU -> fp32 out
  const int aoa_blocks = (Bb * Tt / 16) * (Dd / 16) / 8;   // 8192
  aoa_kernel<<<aoa_blocks, 256, 0, stream>>>(xb, qn, Waoa, baoa, out);
}
